// AudioEncoder_14800457302115
// MI455X (gfx1250) — matrix operand order repaired and checked by reading, not compiled, not run on hardware
//
#include <hip/hip_runtime.h>

#define B_N 4
#define T_N 480000
#define S_N 240000            // (T + 2*2 - 5)/2 + 1
#define STILES (S_N / 16)     // 15000 s-tiles per batch (multiple of TILES_PER_WAVE)
#define TILES_TOTAL (B_N * STILES)   // 60000
#define TILES_PER_WAVE 4
#define WAVES_PER_BLOCK 8

typedef __attribute__((ext_vector_type(16))) __bf16 bf16x16;
typedef __attribute__((ext_vector_type(8)))  float  f32x8;

union Frag {
    bf16x16 v;
    uint4   q[2];
};

__device__ __forceinline__ unsigned short f2bf(float f) {
    unsigned int u = __float_as_uint(f);
    // round-to-nearest-even truncation to bf16
    u += 0x7FFFu + ((u >> 16) & 1u);
    return (unsigned short)(u >> 16);
}

#if defined(__has_builtin) && __has_builtin(__builtin_amdgcn_cvt_pk_bf16_f32)
typedef __attribute__((ext_vector_type(2))) __bf16 bf16x2;
__device__ __forceinline__ unsigned pack_bf16(float a, float b) {
    union { bf16x2 v; unsigned u; } cv;
    cv.v = __builtin_amdgcn_cvt_pk_bf16_f32(a, b);
    return cv.u;
}
#else
__device__ __forceinline__ unsigned pack_bf16(float a, float b) {
    return (unsigned)f2bf(a) | ((unsigned)f2bf(b) << 16);
}
#endif

__device__ __forceinline__ void sched_fence() {
#if defined(__has_builtin) && __has_builtin(__builtin_amdgcn_sched_barrier)
    __builtin_amdgcn_sched_barrier(0);   // stop cross-tile pipelining (VGPR pressure)
#endif
}

__global__ __launch_bounds__(256)
__attribute__((amdgpu_waves_per_eu(4)))
void AudioEncoder_fused_kernel(const float* __restrict__ audio,
                               const float* __restrict__ conv_w,
                               const float* __restrict__ conv_b,
                               const float* __restrict__ lin_w,
                               const float* __restrict__ lin_b,
                               float* __restrict__ out) {
    __shared__ __align__(16) unsigned short lw[64 * 64];   // lin_w bf16 [p][e] (B-frag staging)
    __shared__ __align__(16) float          wconv[64][8];  // [e]: w0..w4, bias, pad, pad

    const int tid = threadIdx.x;

    // ---- one-time block staging ----
    #pragma unroll
    for (int j = 0; j < 16; ++j) {
        int i = tid + j * 256;                 // 4096 elements
        lw[i] = f2bf(lin_w[i]);
    }
    if (tid < 64) {
        #pragma unroll
        for (int k = 0; k < 5; ++k) wconv[tid][k] = conv_w[tid * 5 + k];
        wconv[tid][5] = conv_b[tid];
        wconv[tid][6] = 0.0f;
        wconv[tid][7] = 0.0f;
    }
    __syncthreads();

    const int lane = tid & 31;
    const int wv   = __builtin_amdgcn_readfirstlane(tid >> 5);
    const int row  = lane & 15;   // p-row for B frags; m (s-offset) for A frags / conv
    const int half = lane >> 4;   // 0: lanes 0-15, 1: lanes 16-31
    const int ko   = half * 8;    // A-fragment K base offset for this half-wave

    // ---- per-wave constant B fragments (lin_w^T 32x16 bf16 chunks), read once ----
    // B layout (32x16, wave32): lanes 0-15 hold K=0..7,16..23 (N=lane), lanes 16-31 K=8..15,24..31.
    Frag bfrag[4][2];
    #pragma unroll
    for (int nt = 0; nt < 4; ++nt) {
        #pragma unroll
        for (int kk = 0; kk < 2; ++kk) {
            const unsigned short* src = &lw[(nt * 16 + row) * 64 + kk * 32 + half * 8];
            bfrag[nt][kk].q[0] = *(const uint4*)(src + 0);   // K +0..7   -> VGPR 0-3
            bfrag[nt][kk].q[1] = *(const uint4*)(src + 16);  // K +16..23 -> VGPR 4-7
        }
    }
    float bias[4];
    #pragma unroll
    for (int nt = 0; nt < 4; ++nt) bias[nt] = lin_b[nt * 16 + row];

    const int gwave = blockIdx.x * WAVES_PER_BLOCK + wv;
    const int tile0 = gwave * TILES_PER_WAVE;
    if (tile0 >= TILES_TOTAL) return;            // after __syncthreads: safe

    // STILES % TILES_PER_WAVE == 0 -> a wave's tiles never cross a batch boundary
    const int bb = tile0 / STILES;
    int s0 = (tile0 - bb * STILES) * 16;

    const float* __restrict__ xrow  = audio + (long long)bb * T_N;
    float* __restrict__       obase = out + (long long)bb * S_N * 64;
    const int m = row;

    for (int t = 0; t < TILES_PER_WAVE; ++t, s0 += 16) {
        // ---- conv + ReLU: lane computes EXACTLY the 32 channels its A-fragments
        //      need (K runs {ko..ko+7, ko+16..23, ko+32..39, ko+48..55}), packs them
        //      straight into A-fragment VGPRs. All lane-local: no LDS, no race. ----
        float xv[5];
        #pragma unroll
        for (int k = 0; k < 5; ++k) {
            const int xi = 2 * (s0 + m) - 2 + k;
            xv[k] = ((unsigned)xi < (unsigned)T_N) ? xrow[xi] : 0.0f;
        }

        Frag af[2];
        #pragma unroll
        for (int run = 0; run < 4; ++run) {
            const int ebase = ko + (run & 1) * 16 + (run >> 1) * 32;
            unsigned dw[4];
            #pragma unroll
            for (int jj = 0; jj < 8; jj += 2) {
                const int e = ebase + jj;
                const float4 wa = *(const float4*)&wconv[e][0];
                const float2 wb = *(const float2*)&wconv[e][4];     // w4, bias
                const float4 wc = *(const float4*)&wconv[e + 1][0];
                const float2 wd = *(const float2*)&wconv[e + 1][4];
                float fa = wb.y;
                fa = fmaf(xv[0], wa.x, fa); fa = fmaf(xv[1], wa.y, fa);
                fa = fmaf(xv[2], wa.z, fa); fa = fmaf(xv[3], wa.w, fa);
                fa = fmaf(xv[4], wb.x, fa);
                float fb = wd.y;
                fb = fmaf(xv[0], wc.x, fb); fb = fmaf(xv[1], wc.y, fb);
                fb = fmaf(xv[2], wc.z, fb); fb = fmaf(xv[3], wc.w, fb);
                fb = fmaf(xv[4], wd.x, fb);
                fa = fmaxf(fa, 0.0f);
                fb = fmaxf(fb, 0.0f);
                dw[jj >> 1] = pack_bf16(fa, fb);   // K=even in [15:0], odd in [31:16]
            }
            uint4 q;
            q.x = dw[0]; q.y = dw[1]; q.z = dw[2]; q.w = dw[3];
            af[run >> 1].q[run & 1] = q;   // run0->af[0].q0, run1->af[0].q1, run2->af[1].q0, run3->af[1].q1
        }

        // ---- 8x WMMA: 16(s) x 64(p) f32 tile, acc seeded with lin_b ----
        f32x8 acc[4];
        #pragma unroll
        for (int nt = 0; nt < 4; ++nt) {
            f32x8 c;
            #pragma unroll
            for (int r = 0; r < 8; ++r) c[r] = bias[nt];
            c = __builtin_amdgcn_wmma_f32_16x16x32_bf16(
                    false, af[0].v, false, bfrag[nt][0].v, (short)0, c, false, false);
            c = __builtin_amdgcn_wmma_f32_16x16x32_bf16(
                    false, af[1].v, false, bfrag[nt][1].v, (short)0, c, false, false);
            acc[nt] = c;
        }

        // ---- store: lane owns p = row+16*nt, s = s0 + half*8 + r ----
        float* orow = obase + (long long)(s0 + half * 8) * 64 + row;
        #pragma unroll
        for (int nt = 0; nt < 4; ++nt) {
            #pragma unroll
            for (int r = 0; r < 8; ++r) {
                orow[r * 64 + nt * 16] = acc[nt][r];
            }
        }
        sched_fence();
    }
}

extern "C" void kernel_launch(void* const* d_in, const int* in_sizes, int n_in,
                              void* d_out, int out_size, void* d_ws, size_t ws_size,
                              hipStream_t stream) {
    (void)in_sizes; (void)n_in; (void)out_size; (void)d_ws; (void)ws_size;
    const float* audio  = (const float*)d_in[0];   // (4, 480000) f32
    const float* conv_w = (const float*)d_in[1];   // (64, 1, 5)  f32
    const float* conv_b = (const float*)d_in[2];   // (64,)       f32
    const float* lin_w  = (const float*)d_in[3];   // (64, 64)    f32
    const float* lin_b  = (const float*)d_in[4];   // (64,)       f32
    float* out = (float*)d_out;                    // (4, 240000, 64) f32

    const int waves  = (TILES_TOTAL + TILES_PER_WAVE - 1) / TILES_PER_WAVE; // 15000
    const int blocks = (waves + WAVES_PER_BLOCK - 1) / WAVES_PER_BLOCK;     // 1875
    AudioEncoder_fused_kernel<<<blocks, 256, 0, stream>>>(
        audio, conv_w, conv_b, lin_w, lin_b, out);
}
